// KimiK25TextMoEGate_55662776156364
// MI455X (gfx1250) — matrix operand order repaired and checked by reading, not compiled, not run on hardware
//
#include <hip/hip_runtime.h>
#include <hip/hip_bf16.h>

typedef __attribute__((ext_vector_type(16))) __bf16 v16bf;
typedef __attribute__((ext_vector_type(8)))  __bf16 v8bf;
typedef __attribute__((ext_vector_type(4)))  __bf16 v4bf;
typedef __attribute__((ext_vector_type(8)))  float  v8f;
typedef __attribute__((ext_vector_type(4)))  float  v4f;

#define HID     7168
#define NEXP    256
#define NGRP    8
#define EPG     32      // experts per group
#define TOPKG   4
#define TOPK    8
#define MTILE   32      // tokens per workgroup
#define KC      512     // K-chunk staged in LDS (power of 2, divides HID)
#define LDA     (KC + 8)  // bf16 row stride in LDS: bank stride 4 dwords -> conflict-free
#define NTHREADS 256
#define SCALING 2.5f

// LDS: A-stage (32 x 520 bf16 = 33280B) aliased by scores (32 x 256 f32 = 32768B),
// plus group-score scratch (32 x 8 f32 = 1024B). Total 34304B.
#define SMEM_A_BYTES (MTILE * LDA * 2)
#define SMEM_BYTES   (SMEM_A_BYTES + MTILE * NGRP * 4)

// ---------------------------------------------------------------------------
// Prepass: pack fp32 weights -> bf16 in d_ws (run once per launch; weights are
// re-read by every workgroup, so packing halves weight-side L2 traffic and
// removes 8 v_cvt_pk per B-fragment from the GEMM hot loop).
// ---------------------------------------------------------------------------
__global__ __launch_bounds__(256)
void wpack_kernel(const float* __restrict__ w, __bf16* __restrict__ wbf)
{
    size_t i = ((size_t)blockIdx.x * 256 + threadIdx.x) * 8;  // grid sized exactly
    v4f a = *(const v4f*)(w + i);
    v4f b = *(const v4f*)(w + i + 4);
    v8bf o;
    o[0] = (__bf16)a[0]; o[1] = (__bf16)a[1]; o[2] = (__bf16)a[2]; o[3] = (__bf16)a[3];
    o[4] = (__bf16)b[0]; o[5] = (__bf16)b[1]; o[6] = (__bf16)b[2]; o[7] = (__bf16)b[3];
    *(v8bf*)(wbf + i) = o;
}

// ---------------------------------------------------------------------------
// Main fused gate kernel. PACKED=true reads bf16 weights from d_ws.
// ---------------------------------------------------------------------------
template<bool PACKED>
__global__ __launch_bounds__(NTHREADS)
void moe_gate_kernel(const float*  __restrict__ x,     // [T, HID] fp32
                     const float*  __restrict__ w,     // [NEXP, HID] fp32
                     const __bf16* __restrict__ wbf,   // [NEXP, HID] bf16 (PACKED path)
                     const float*  __restrict__ bias,  // [NEXP]
                     int*   __restrict__ out_idx,      // [T, TOPK]
                     float* __restrict__ out_wt,       // [T, TOPK]
                     int T)
{
    __shared__ __align__(16) unsigned char smem_raw[SMEM_BYTES];
    __bf16* As      = (__bf16*)smem_raw;
    float*  scoresS = (float*)smem_raw;                    // reused after GEMM
    float*  gsc     = (float*)(smem_raw + SMEM_A_BYTES);   // [MTILE][NGRP]

    const int tid     = threadIdx.x;
    const int wave    = tid >> 5;
    const int lane    = tid & 31;
    const int tokBase = blockIdx.x * MTILE;
    const int laneN   = lane & 15;       // matrix row (A) / column (B) within 16
    const int hiHalf  = lane >> 4;       // 0 or 1
    const int aOff    = hiHalf ? 16 : 0; // A K-offset for this lane half
    const int bOff    = hiHalf << 4;     // B K-offset (0 or 16)
    const int eBase   = wave * 32;       // this wave's 32 experts

    v8f acc[2][2] = {};  // [mTile][nTile] f32 accumulators

    for (int k0 = 0; k0 < HID; k0 += KC) {
        __syncthreads();  // protect LDS reuse across chunks
        // Cooperative stage: x[tokBase..+31][k0..k0+KC) -> bf16 in LDS
        for (int i = tid * 4; i < MTILE * KC; i += NTHREADS * 4) {
            int row = i >> 9;           // i / KC  (KC == 512)
            int col = i & (KC - 1);
            v4f v = *(const v4f*)(x + (size_t)(tokBase + row) * HID + k0 + col);
            v4bf b;
            b[0] = (__bf16)v[0]; b[1] = (__bf16)v[1];
            b[2] = (__bf16)v[2]; b[3] = (__bf16)v[3];
            *(v4bf*)(As + row * LDA + col) = b;
        }
        __syncthreads();

        for (int kk = 0; kk < KC; kk += 32) {
            // B fragments (weights): 16 contiguous K values per lane.
            v16bf bf[2];
#pragma unroll
            for (int nt = 0; nt < 2; ++nt) {
                const size_t erow = (size_t)(eBase + nt * 16 + laneN) * HID
                                    + (size_t)(k0 + kk + bOff);
                if (PACKED) {
                    const __bf16* wp = wbf + erow;
                    v8bf lo = *(const v8bf*)wp;        // K {off..off+7}
                    v8bf hi = *(const v8bf*)(wp + 8);  // K {off+8..off+15}
                    bf[nt] = __builtin_shufflevector(lo, hi,
                             0,1,2,3,4,5,6,7,8,9,10,11,12,13,14,15);
                } else {
                    const float* wp = w + erow;
                    v16bf t;
#pragma unroll
                    for (int j = 0; j < 16; ++j) t[j] = (__bf16)wp[j];
                    bf[nt] = t;
                }
            }
            // A fragments from LDS (already bf16): two 16B ds_load_b128 per tile
#pragma unroll
            for (int mt = 0; mt < 2; ++mt) {
                const __bf16* ap = As + (mt * 16 + laneN) * LDA + kk + aOff;
                v8bf lo = *(const v8bf*)ap;          // K {off..off+7}
                v8bf hi = *(const v8bf*)(ap + 8);    // K {off+8..off+15}
                v16bf a = __builtin_shufflevector(lo, hi,
                          0,1,2,3,4,5,6,7,8,9,10,11,12,13,14,15);
                acc[mt][0] = __builtin_amdgcn_wmma_f32_16x16x32_bf16(
                                 false, a, false, bf[0], (short)0, acc[mt][0], false, false);
                acc[mt][1] = __builtin_amdgcn_wmma_f32_16x16x32_bf16(
                                 false, a, false, bf[1], (short)0, acc[mt][1], false, false);
            }
        }
    }

    __syncthreads();
    // Scatter accumulators to scores LDS. C/D layout: VGPR j, lanes0-15: M=j, N=lane;
    // lanes16-31: M=j+8, N=lane-16.
#pragma unroll
    for (int mt = 0; mt < 2; ++mt)
#pragma unroll
        for (int nt = 0; nt < 2; ++nt)
#pragma unroll
            for (int j = 0; j < 8; ++j) {
                int tl = mt * 16 + j + (hiHalf << 3);
                int e  = eBase + nt * 16 + laneN;
                scoresS[tl * NEXP + e] = acc[mt][nt][j];
            }
    __syncthreads();

    // Phase A: sigmoid + bias (in place), group top-2 sum. 8 threads/token, 1 group each.
    {
        int tl = tid >> 3;
        int g  = tid & 7;
        float m1 = -1e30f, m2 = -1e30f;
        int ebase = g * EPG;
#pragma unroll 1
        for (int j = 0; j < EPG; ++j) {
            int e = ebase + j;
            float raw = scoresS[tl * NEXP + e];
            float s = 1.0f / (1.0f + __expf(-raw));
            float v = s + bias[e];
            scoresS[tl * NEXP + e] = v;  // store biased score
            if (v > m1) { m2 = m1; m1 = v; }
            else if (v > m2) { m2 = v; }
        }
        gsc[tl * NGRP + g] = m1 + m2;
    }
    __syncthreads();

    // Phase B: per-token top-4 groups, then masked top-8 experts.
    if (tid < MTILE) {
        int tl = tid;
        unsigned gmask = 0;
#pragma unroll
        for (int r = 0; r < TOPKG; ++r) {
            float best = -1e30f; int bg = 0;
#pragma unroll
            for (int g = 0; g < NGRP; ++g)
                if (!((gmask >> g) & 1u)) {
                    float v = gsc[tl * NGRP + g];
                    if (v > best) { best = v; bg = g; }
                }
            gmask |= 1u << bg;
        }
        int sel[TOPKG]; int ns = 0;
        for (int g = 0; g < NGRP; ++g)
            if ((gmask >> g) & 1u) { if (ns < TOPKG) sel[ns++] = g; }

        unsigned chosen[TOPKG] = {0u, 0u, 0u, 0u};
        int idxs[TOPK]; float wts[TOPK]; float wsum = 0.0f;
#pragma unroll 1
        for (int r = 0; r < TOPK; ++r) {
            float best = -1e30f; int bq = 0, bj = 0;
#pragma unroll 1
            for (int q = 0; q < TOPKG; ++q) {
                int base = sel[q] * EPG;
#pragma unroll 1
                for (int j = 0; j < EPG; ++j) {
                    if ((chosen[q] >> j) & 1u) continue;
                    float v = scoresS[tl * NEXP + base + j];
                    if (v > best) { best = v; bq = q; bj = j; }
                }
            }
            chosen[bq] |= 1u << bj;
            int e = sel[bq] * EPG + bj;
            idxs[r] = e;
            float wv = best - bias[e];   // recover sigmoid score (without bias)
            wts[r] = wv; wsum += wv;
        }
        float scale = SCALING / (wsum + 1e-20f);
        long long tok = (long long)tokBase + tl;
#pragma unroll
        for (int r = 0; r < TOPK; ++r) {
            out_idx[tok * TOPK + r] = idxs[r];
            out_wt [tok * TOPK + r] = wts[r] * scale;
        }
    }
}

extern "C" void kernel_launch(void* const* d_in, const int* in_sizes, int n_in,
                              void* d_out, int out_size, void* d_ws, size_t ws_size,
                              hipStream_t stream)
{
    const float* x    = (const float*)d_in[0];   // [4,4096,7168] fp32
    const float* w    = (const float*)d_in[1];   // [256,7168] fp32
    const float* bias = (const float*)d_in[2];   // [256] fp32

    int T = in_sizes[0] / HID;                   // 16384
    int*   out_idx = (int*)d_out;                        // first output: topk_idx [T,8]
    float* out_wt  = (float*)d_out + (size_t)T * TOPK;   // second output: topk_weight [T,8]

    int nblocks = T / MTILE;                     // T = 16384 divisible by 32

    const size_t wbf_bytes = (size_t)NEXP * HID * sizeof(__bf16);  // 3.67 MB
    if (ws_size >= wbf_bytes) {
        __bf16* wbf = (__bf16*)d_ws;
        wpack_kernel<<<(NEXP * HID) / (256 * 8), 256, 0, stream>>>(w, wbf);
        moe_gate_kernel<true><<<nblocks, NTHREADS, 0, stream>>>(
            x, w, wbf, bias, out_idx, out_wt, T);
    } else {
        moe_gate_kernel<false><<<nblocks, NTHREADS, 0, stream>>>(
            x, w, (const __bf16*)nullptr, bias, out_idx, out_wt, T);
    }
}
